// PlaneSweepCostVolume_81149112090846
// MI455X (gfx1250) — matrix operand order repaired and checked by reading, not compiled, not run on hardware
//
#include <hip/hip_runtime.h>

// Plane-sweep cost volume (NCC over C=16 channels) for MI455X / gfx1250.
// Wave32-cooperative: each wave owns a 16-pixel strip for one (b, depth).
// NCC channel reductions via v_wmma_f32_16x16x32_f16:
//   A(16x32)  = [W | R~]   (rows = pixels, f16)
//   B1(32x16) = [W^T ; 0]  -> diag = sum_c w^2   (a2)
//   B2(32x16) = [0 ; W^T]  -> diag = sum_c r~*w  (a3)
// B zero-halves come from a zeroed LDS region (no per-element selects).
// a1 = sum_c w and s_r = sum_c r~^2 via local sums + shfl_xor(16).
// Uniform camera params are forced onto the scalar path via readfirstlane.

#define B_ 2
#define C_ 16
#define H_ 160
#define W_ 192
#define S_ 3
#define D_ 48
#define HW_ (H_ * W_)
#define EPS_ 1e-6f
#define WPB 8   // waves per block

typedef __attribute__((ext_vector_type(16))) _Float16 v16h;
typedef __attribute__((ext_vector_type(8)))  float    v8f;

__global__ __launch_bounds__(256) void psv_ncc_kernel(
    const float* __restrict__ refF,    // [B,C,H,W]
    const float* __restrict__ srcF,    // [S,B,C,H,W]
    const float* __restrict__ refK,    // [B,3,3]
    const float* __restrict__ srcK,    // [S,B,3,3]
    const float* __restrict__ Tmat,    // [S,B,4,4]
    const float* __restrict__ depths,  // [D]
    float* __restrict__ out)           // [B,D,H,W]
{
    const int lane = threadIdx.x & 31;
    const int wv   = threadIdx.x >> 5;
    // Force wave-uniform tile id so all camera-parameter loads scalarize.
    const int tile = __builtin_amdgcn_readfirstlane(blockIdx.x * WPB + wv);

    const int TPR = W_ / 16;          // tiles per row = 12
    const int TPI = HW_ / 16;         // tiles per image = 1920
    const int TOTAL = B_ * D_ * TPI;  // 184320
    if (tile >= TOTAL) return;        // scalar-uniform exit

    const int b   = tile / (D_ * TPI);
    const int r0  = tile - b * (D_ * TPI);
    const int d   = r0 / TPI;
    const int ti  = r0 - d * TPI;
    const int row = ti / TPR;
    const int col0 = (ti - row * TPR) * 16;

    const int p  = lane & 15;   // pixel within strip
    const int cb = lane >> 4;   // channel half: 0 -> c0..7, 1 -> c8..15
    const bool lo = (lane < 16);

    // B staging: rows 0..15 = W (pixels x channels), rows 16..31 = zeros.
    __shared__ _Float16 ldsB[WPB][32][16];
    __shared__ _Float16 ldsR[WPB][16][16];
    __shared__ float    ldsD2[WPB][16];
    __shared__ float    ldsD3[WPB][16];

    // Zero the pad rows once (each lane covers 8 contiguous halves).
#pragma unroll
    for (int i = 0; i < 8; i++) ldsB[wv][16 + p][cb*8 + i] = (_Float16)0.0f;

    // ---- invert ref intrinsics (general 3x3 adjugate, scalar path) ----
    const float* Kr = refK + b * 9;
    float a = Kr[0], bk = Kr[1], c = Kr[2];
    float dk = Kr[3], e = Kr[4], f = Kr[5];
    float g = Kr[6], h = Kr[7], i9 = Kr[8];
    float det = a*(e*i9 - f*h) - bk*(dk*i9 - f*g) + c*(dk*h - e*g);
    float id  = 1.0f / det;
    float i00=(e*i9-f*h)*id, i01=(c*h-bk*i9)*id, i02=(bk*f-c*e)*id;
    float i10=(f*g-dk*i9)*id, i11=(a*i9-c*g)*id, i12=(c*dk-a*f)*id;
    float i20=(dk*h-e*g)*id,  i21=(bk*g-a*h)*id, i22=(a*e-bk*dk)*id;

    const float depth = depths[d];
    const float u = (float)(col0 + p);
    const float v = (float)row;
    const float Xx = depth * (i00*u + i01*v + i02);
    const float Xy = depth * (i10*u + i11*v + i12);
    const float Xz = depth * (i20*u + i21*v + i22);

    // ---- reference features: center over channels, compute s_r ----
    const float* rbase = refF + (size_t)b * C_ * HW_ + row * W_ + col0 + p;
    float rf[8];
#pragma unroll
    for (int i = 0; i < 8; i++) rf[i] = rbase[(size_t)(cb*8 + i) * HW_];
    float sum = 0.0f;
#pragma unroll
    for (int i = 0; i < 8; i++) sum += rf[i];
    sum += __shfl_xor(sum, 16);
    const float mean = sum * (1.0f / 16.0f);
    float sr = 0.0f;
#pragma unroll
    for (int i = 0; i < 8; i++) { float t = rf[i] - mean; sr += t*t; rf[i] = t; }
    sr += __shfl_xor(sr, 16);
#pragma unroll
    for (int i = 0; i < 8; i++) ldsR[wv][p][cb*8 + i] = (_Float16)rf[i];

    float acc = 0.0f;

    for (int s = 0; s < S_; s++) {
        const float* Tm = Tmat + ((size_t)s * B_ + b) * 16;   // scalar loads
        const float* Ks = srcK + ((size_t)s * B_ + b) * 9;    // scalar loads

        float Xsx = Tm[0]*Xx + Tm[1]*Xy + Tm[2]*Xz  + Tm[3];
        float Xsy = Tm[4]*Xx + Tm[5]*Xy + Tm[6]*Xz  + Tm[7];
        float Xsz = Tm[8]*Xx + Tm[9]*Xy + Tm[10]*Xz + Tm[11];
        float px = Ks[0]*Xsx + Ks[1]*Xsy + Ks[2]*Xsz;
        float py = Ks[3]*Xsx + Ks[4]*Xsy + Ks[5]*Xsz;
        float pz = Ks[6]*Xsx + Ks[7]*Xsy + Ks[8]*Xsz;
        const bool  valid = pz > 0.001f;
        const float zs = fmaxf(pz, 0.001f);
        const float x = px / zs, y = py / zs;
        const float x0f = floorf(x), y0f = floorf(y);
        const float fx = x - x0f, fy = y - y0f;
        const int xi = (int)x0f, yi = (int)y0f;

        // Branchless bilinear: clamp indices, zero weights when OOB/invalid
        // (matches the reference's clip+masked-weight semantics exactly).
        int   xc[2], yc[2];
        float wx[2], wy[2];
#pragma unroll
        for (int t = 0; t < 2; t++) {
            int xx = xi + t, yy = yi + t;
            bool okx = (xx >= 0) & (xx < W_);
            bool oky = (yy >= 0) & (yy < H_);
            xc[t] = min(max(xx, 0), W_ - 1);
            yc[t] = min(max(yy, 0), H_ - 1);
            wx[t] = (okx && valid) ? (t ? fx : 1.0f - fx) : 0.0f;
            wy[t] = oky ? (t ? fy : 1.0f - fy) : 0.0f;
        }

        const float* sb = srcF + ((size_t)s * B_ + b) * C_ * HW_
                               + (size_t)cb * 8 * HW_;
        const int o00 = yc[0] * W_ + xc[0];
        const int o01 = yc[0] * W_ + xc[1];
        const int o10 = yc[1] * W_ + xc[0];
        const int o11 = yc[1] * W_ + xc[1];
        const float w00 = wx[0] * wy[0], w01 = wx[1] * wy[0];
        const float w10 = wx[0] * wy[1], w11 = wx[1] * wy[1];

        float w[8];
#pragma unroll
        for (int i = 0; i < 8; i++) {
            const float* ch = sb + (size_t)i * HW_;
            w[i] = ch[o00] * w00 + ch[o01] * w01
                 + ch[o10] * w10 + ch[o11] * w11;
        }

        float a1 = 0.0f;
#pragma unroll
        for (int i = 0; i < 8; i++) a1 += w[i];
        a1 += __shfl_xor(a1, 16);
#pragma unroll
        for (int i = 0; i < 8; i++) ldsB[wv][p][cb*8 + i] = (_Float16)w[i];
        __builtin_amdgcn_wave_barrier();   // in-wave LDS ops are in order

        // ---- fragments in contiguous K order (k = 16h + j) for A and B ----
        // A 16x32: lanes 0-15 hold M=lane, K=0..15; lanes 16-31 K=16..31
        v16h A, B1, B2;
        const int hiK = lo ? 0 : 16;
#pragma unroll
        for (int j = 0; j < 16; j++) {
            int K = j + hiK;
            A[j] = (K < 16) ? ldsB[wv][p][K] : ldsR[wv][p][K - 16];
        }
        // B 32x16: lane N=lane&15; lanes 0-15 hold K=0..15, lanes 16-31 K=16..31
        // Zero halves come from pad rows 16..31 of ldsB (no VALU selects).
        const int rB1 = p + (lo ? 0 : 16);
        const int rB2 = p + (lo ? 16 : 0);
#pragma unroll
        for (int j = 0; j < 16; j++) {
            B1[j] = ldsB[wv][rB1][j];
            B2[j] = ldsB[wv][rB2][j];
        }

        v8f cz = {0.f,0.f,0.f,0.f,0.f,0.f,0.f,0.f};
        v8f c1 = __builtin_amdgcn_wmma_f32_16x16x32_f16(
                     false, A, false, B1, (short)0, cz, false, false); // diag=a2
        v8f c2 = __builtin_amdgcn_wmma_f32_16x16x32_f16(
                     false, A, false, B2, (short)0, cz, false, false); // diag=a3

        // diag(M=N): lanes 0-7 elem=lane; lanes 24-31 elem=lane-24
        const int rd = p - (lo ? 0 : 8);
#pragma unroll
        for (int r = 0; r < 8; r++) {
            if (rd == r) { ldsD2[wv][p] = c1[r]; ldsD3[wv][p] = c2[r]; }
        }
        __builtin_amdgcn_wave_barrier();

        const float a2 = ldsD2[wv][p];
        const float a3 = ldsD3[wv][p];
        const float varw = fmaxf(a2 - a1 * a1 * (1.0f / 16.0f), 0.0f);
        const float den  = sqrtf(sr) * sqrtf(varw) + EPS_;
        acc += a3 / den;
        __builtin_amdgcn_wave_barrier();
    }

    if (lo) {
        out[(((size_t)b * D_ + d) * H_ + row) * W_ + col0 + p] =
            acc * (1.0f / (float)S_);
    }
}

extern "C" void kernel_launch(void* const* d_in, const int* in_sizes, int n_in,
                              void* d_out, int out_size, void* d_ws, size_t ws_size,
                              hipStream_t stream) {
    const float* refF   = (const float*)d_in[0];
    const float* srcF   = (const float*)d_in[1];
    const float* refK   = (const float*)d_in[2];
    const float* srcK   = (const float*)d_in[3];
    const float* Tmat   = (const float*)d_in[4];
    const float* depths = (const float*)d_in[5];
    float* out = (float*)d_out;

    const int tiles  = B_ * D_ * (HW_ / 16);    // 184320 wave-tiles
    const int blocks = (tiles + WPB - 1) / WPB; // 23040 blocks of 8 waves
    hipLaunchKernelGGL(psv_ncc_kernel, dim3(blocks), dim3(256), 0, stream,
                       refF, srcF, refK, srcK, Tmat, depths, out);
}
